// My_GNN_GNN_NN_28647431864457
// MI455X (gfx1250) — matrix operand order repaired and checked by reading, not compiled, not run on hardware
//
#include <hip/hip_runtime.h>
#include <hip/hip_bf16.h>

typedef __attribute__((ext_vector_type(16))) _Float16 v16h;
typedef __attribute__((ext_vector_type(8)))  float    v8f;
typedef __attribute__((ext_vector_type(4)))  unsigned int u32x4;

#define NGRAPH 1000
#define NBUS   30
#define NEDGE  34
#define HID    3000
#define HIDP   3008         // padded hidden dim (multiple of 32)
#define KTILES (HIDP / 32)  // 94

static __device__ __forceinline__ v8f wmma_f16(v16h a, v16h b, v8f c) {
  // (neg_a, A, neg_b, B, c_mod, C, reuse_a, reuse_b)
  return __builtin_amdgcn_wmma_f32_16x16x32_f16(false, a, false, b, (short)0, c, false, false);
}

// Hardware tanh (CDNA5 V_TANH_F32 transcendental; co-executes with WMMA).
static __device__ __forceinline__ float fast_tanh(float x) {
#if __has_builtin(__builtin_amdgcn_tanhf)
  return __builtin_amdgcn_tanhf(x);
#else
  float y;
  asm volatile("v_tanh_f32 %0, %1\n\tv_nop\n\tv_nop" : "=v"(y) : "v"(x));
  return y;
#endif
}

// Load a 16x32 f16 A-fragment (M = mt*16 + lane%16) from an LDS 32x32 f16 tile
// (row-major, 64B row stride). Contiguous K order: lanes 0-15 -> K 0..15 (bytes 0..31);
// lanes 16-31 -> K 16..31 (bytes 32..63).
static __device__ __forceinline__ v16h load_afrag(const _Float16* Lz, int lane, int mt) {
  const int m    = mt * 16 + (lane & 15);
  const int koff = (lane < 16) ? 0 : 32;   // bytes
  const char* base = (const char*)Lz + m * 64 + koff;
  v16h a;
  *((u32x4*)&a)     = *(const u32x4*)(base);
  *((u32x4*)&a + 1) = *(const u32x4*)(base + 16);
  return a;
}

// ---------------- prep: per-graph G = A_hat @ tanh(A_hat @ (X W1) + b1) ----------------
__global__ __launch_bounds__(32) void prep_graph(const float* __restrict__ x,
                                                 const int* __restrict__ ei,
                                                 const float* __restrict__ W1,
                                                 const float* __restrict__ b1,
                                                 _Float16* __restrict__ Gf16) {
  __shared__ float sxw[NBUS * 8];
  __shared__ float sh1[NBUS * 8];
  __shared__ float sdinv[NBUS];
  __shared__ int   sr[NEDGE];
  __shared__ int   sc[NEDGE];
  const int g = blockIdx.x;
  const int t = threadIdx.x;

  // topology identical for all graphs: use graph 0's edges (values 0..29)
  for (int e = t; e < NEDGE; e += 32) {
    sr[e] = ei[e];
    sc[e] = ei[NEDGE * NGRAPH + e];
  }
  __syncthreads();

  if (t < NBUS) {
    float d = 1.0f;  // self loop
    for (int e = 0; e < NEDGE; ++e) d += (sc[e] == t) ? 1.0f : 0.0f;
    sdinv[t] = rsqrtf(d);
    const float* xr = x + (size_t)(g * NBUS + t) * 4;
    const float x0 = xr[0], x1 = xr[1], x2 = xr[2], x3 = xr[3];
    for (int j = 0; j < 8; ++j)
      sxw[t * 8 + j] = x0 * W1[0 * 8 + j] + x1 * W1[1 * 8 + j] +
                       x2 * W1[2 * 8 + j] + x3 * W1[3 * 8 + j];
  }
  __syncthreads();

  if (t < NBUS) {
    const float dm = sdinv[t];
    for (int j = 0; j < 8; ++j) {
      float a = dm * dm * sxw[t * 8 + j];
      for (int e = 0; e < NEDGE; ++e)
        if (sc[e] == t) a += sdinv[sr[e]] * dm * sxw[sr[e] * 8 + j];
      sh1[t * 8 + j] = fast_tanh(a + b1[j]);
    }
  }
  __syncthreads();

  if (t < NBUS) {
    const float dm = sdinv[t];
    for (int j = 0; j < 8; ++j) {
      float a = dm * dm * sh1[t * 8 + j];
      for (int e = 0; e < NEDGE; ++e)
        if (sc[e] == t) a += sdinv[sr[e]] * dm * sh1[sr[e] * 8 + j];
      Gf16[(size_t)(g * 32 + t) * 8 + j] = (_Float16)a;
    }
  } else {  // pad rows 30,31 with zeros
    for (int j = 0; j < 8; ++j) Gf16[(size_t)(g * 32 + t) * 8 + j] = (_Float16)0.f;
  }
}

// ---------------- prep: f16 transposed / padded weights + padded biases ----------------
// W2T3: [HIDP][32] halves; [c][0..7] = W2[:,c], rest zero -> both lane halves of the
// B-fragment load unconditionally (hi half fetches zeros, matching K=8..31 == 0).
__global__ void prep_w2t(const float* __restrict__ W2, const float* __restrict__ b2,
                         _Float16* __restrict__ W2T3, float* __restrict__ b2p) {
  const int c = blockIdx.x * blockDim.x + threadIdx.x;
  if (c >= HIDP) return;
  for (int k = 0; k < 8; ++k)
    W2T3[c * 32 + k] = (c < HID) ? (_Float16)W2[k * HID + c] : (_Float16)0.f;
  for (int k = 8; k < 32; ++k) W2T3[c * 32 + k] = (_Float16)0.f;
  b2p[c] = (c < HID) ? b2[c] : 0.f;
}
__global__ void prep_wl1t(const float* __restrict__ Wl1, _Float16* __restrict__ Wl1T) {
  const int idx = blockIdx.x * blockDim.x + threadIdx.x;  // [32][HIDP]
  if (idx >= 32 * HIDP) return;
  const int n = idx / HIDP, k = idx % HIDP;
  Wl1T[idx] = (n < NBUS && k < HID) ? (_Float16)Wl1[k * NBUS + n] : (_Float16)0.f;
}
__global__ void prep_small(const float* __restrict__ Wl2, const float* __restrict__ bl1,
                           const float* __restrict__ bl2, _Float16* __restrict__ Wl2T,
                           float* __restrict__ bl1p, float* __restrict__ bl2p) {
  const int idx = blockIdx.x * blockDim.x + threadIdx.x;
  if (idx < 64 * 32) {
    const int n = idx / 32, o = idx % 32;
    Wl2T[idx] = (n < 60 && o < NBUS) ? (_Float16)Wl2[o * 60 + n] : (_Float16)0.f;
  }
  if (idx < 32) bl1p[idx] = (idx < NBUS) ? bl1[idx] : 0.f;
  if (idx < 64) bl2p[idx] = (idx < 60) ? bl2[idx] : 0.f;
}

// ---------------- main fused kernel: one wave per graph ----------------
__global__ __launch_bounds__(256) void gcn_main(const _Float16* __restrict__ Gf16,
                                                const _Float16* __restrict__ W2T3,
                                                const _Float16* __restrict__ Wl1T,
                                                const _Float16* __restrict__ Wl2T,
                                                const float* __restrict__ b2p,
                                                const float* __restrict__ bl1p,
                                                const float* __restrict__ bl2p,
                                                float* __restrict__ out) {
  __shared__ _Float16 lds[8 * 32 * 32];  // 2KB per wave
  const int lane = threadIdx.x & 31;
  const int wave = threadIdx.x >> 5;
  const int g    = blockIdx.x * 8 + wave;
  _Float16* Lz   = &lds[wave * 1024];
  const bool lo  = lane < 16;
  const int  n   = lane & 15;
  const int  mb  = lo ? 0 : 8;   // C-layout row offset within 16x16 tile
  const int  kh  = lo ? 0 : 16;  // B-layout K offset (halfwords)

  // A-fragments for GEMM1: G (16x32, K=8 valid). lanes 16-31 carry K=16..31 -> zero.
  v16h aG0 = {}; v16h aG1 = {};
  if (lo) {
    *(u32x4*)&aG0 = *(const u32x4*)(Gf16 + (size_t)(g * 32 + n) * 8);
    *(u32x4*)&aG1 = *(const u32x4*)(Gf16 + (size_t)(g * 32 + 16 + n) * 8);
  }

  v8f acc00 = {}, acc01 = {}, acc10 = {}, acc11 = {};

  for (int kt = 0; kt < KTILES; ++kt) {
    const int c0 = kt * 32;

    // ---- GEMM1: Z tile (32 nodes x 32 hidden) = G @ W2[:,c0:c0+32] ----
    const _Float16* pw0 = W2T3 + (size_t)(c0 + n) * 32 + kh;
    const _Float16* pw1 = W2T3 + (size_t)(c0 + 16 + n) * 32 + kh;
    v16h bw0, bw1;
    *((u32x4*)&bw0)     = *(const u32x4*)(pw0);
    *((u32x4*)&bw0 + 1) = *(const u32x4*)(pw0 + 8);
    *((u32x4*)&bw1)     = *(const u32x4*)(pw1);
    *((u32x4*)&bw1 + 1) = *(const u32x4*)(pw1 + 8);

    v8f zf = {};
    v8f z00 = wmma_f16(aG0, bw0, zf);
    v8f z01 = wmma_f16(aG0, bw1, zf);
    v8f z10 = wmma_f16(aG1, bw0, zf);
    v8f z11 = wmma_f16(aG1, bw1, zf);

    const float bias0 = b2p[c0 + n];
    const float bias1 = b2p[c0 + 16 + n];

    // H2 = tanh(Z + b2): C-layout -> LDS [m][k] f16 tile
#pragma unroll
    for (int r = 0; r < 8; ++r) {
      Lz[(mb + r) * 32 + n]            = (_Float16)fast_tanh(z00[r] + bias0);
      Lz[(mb + r) * 32 + 16 + n]       = (_Float16)fast_tanh(z01[r] + bias1);
      Lz[(16 + mb + r) * 32 + n]       = (_Float16)fast_tanh(z10[r] + bias0);
      Lz[(16 + mb + r) * 32 + 16 + n]  = (_Float16)fast_tanh(z11[r] + bias1);
    }
    asm volatile("s_wait_dscnt 0" ::: "memory");  // wave-local LDS visibility

    // ---- GEMM2: acc += H2_tile @ Wl1[c0:c0+32, :] ----
    v16h a0 = load_afrag(Lz, lane, 0);
    v16h a1 = load_afrag(Lz, lane, 1);

    const _Float16* p0 = Wl1T + (size_t)n * HIDP + c0 + kh;
    const _Float16* p1 = Wl1T + (size_t)(16 + n) * HIDP + c0 + kh;
    v16h bb0, bb1;
    *((u32x4*)&bb0)     = *(const u32x4*)(p0);
    *((u32x4*)&bb0 + 1) = *(const u32x4*)(p0 + 8);
    *((u32x4*)&bb1)     = *(const u32x4*)(p1);
    *((u32x4*)&bb1 + 1) = *(const u32x4*)(p1 + 8);

    acc00 = wmma_f16(a0, bb0, acc00);
    acc01 = wmma_f16(a0, bb1, acc01);
    acc10 = wmma_f16(a1, bb0, acc10);
    acc11 = wmma_f16(a1, bb1, acc11);
  }

  // ---- epilogue: H3 = relu(acc + bl1) -> LDS; out = H3 @ Wl2 + bl2 ----
  const float bl10 = bl1p[n];
  const float bl11 = bl1p[16 + n];
#pragma unroll
  for (int r = 0; r < 8; ++r) {
    Lz[(mb + r) * 32 + n]           = (_Float16)fmaxf(acc00[r] + bl10, 0.f);
    Lz[(mb + r) * 32 + 16 + n]      = (_Float16)fmaxf(acc01[r] + bl11, 0.f);
    Lz[(16 + mb + r) * 32 + n]      = (_Float16)fmaxf(acc10[r] + bl10, 0.f);
    Lz[(16 + mb + r) * 32 + 16 + n] = (_Float16)fmaxf(acc11[r] + bl11, 0.f);
  }
  asm volatile("s_wait_dscnt 0" ::: "memory");

  v16h h0 = load_afrag(Lz, lane, 0);
  v16h h1 = load_afrag(Lz, lane, 1);

  v8f oacc[2][4];
#pragma unroll
  for (int j = 0; j < 4; ++j) {
    const _Float16* pw = Wl2T + (size_t)(j * 16 + n) * 32 + kh;
    v16h bw;
    *((u32x4*)&bw)     = *(const u32x4*)(pw);
    *((u32x4*)&bw + 1) = *(const u32x4*)(pw + 8);
    v8f zf = {};
    oacc[0][j] = wmma_f16(h0, bw, zf);
    oacc[1][j] = wmma_f16(h1, bw, zf);
  }

#pragma unroll
  for (int mt = 0; mt < 2; ++mt) {
#pragma unroll
    for (int j = 0; j < 4; ++j) {
      const int col = j * 16 + n;
      const float bb = bl2p[col];
#pragma unroll
      for (int r = 0; r < 8; ++r) {
        const int m = mt * 16 + mb + r;
        if (m < NBUS && col < 60)
          out[((size_t)g * NBUS + m) * 60 + col] = oacc[mt][j][r] + bb;
      }
    }
  }
}

extern "C" void kernel_launch(void* const* d_in, const int* in_sizes, int n_in,
                              void* d_out, int out_size, void* d_ws, size_t ws_size,
                              hipStream_t stream) {
  const float* x   = (const float*)d_in[0];
  const int*   ei  = (const int*)d_in[1];
  const float* W1  = (const float*)d_in[2];
  const float* b1  = (const float*)d_in[3];
  const float* W2  = (const float*)d_in[4];
  const float* b2  = (const float*)d_in[5];
  const float* Wl1 = (const float*)d_in[6];
  const float* bl1 = (const float*)d_in[7];
  const float* Wl2 = (const float*)d_in[8];
  const float* bl2 = (const float*)d_in[9];
  float* out = (float*)d_out;

  char* ws = (char*)d_ws;
  _Float16* Gf16 = (_Float16*)(ws);                    // 1000*32*8*2 = 512000 B
  _Float16* W2T3 = (_Float16*)(ws + 512000);           // 3008*32*2   = 192512 B
  _Float16* Wl1T = (_Float16*)(ws + 704512);           // 32*3008*2   = 192512 B
  _Float16* Wl2T = (_Float16*)(ws + 897024);           // 64*32*2     =   4096 B
  float*    b2p  = (float*)(ws + 901120);              // 3008*4      =  12032 B
  float*    bl1p = (float*)(ws + 913152);              // 32*4        =    128 B
  float*    bl2p = (float*)(ws + 913280);              // 64*4        =    256 B

  prep_w2t <<<(HIDP + 255) / 256, 256, 0, stream>>>(W2, b2, W2T3, b2p);
  prep_wl1t<<<(32 * HIDP + 255) / 256, 256, 0, stream>>>(Wl1, Wl1T);
  prep_small<<<(64 * 32 + 255) / 256, 256, 0, stream>>>(Wl2, bl1, bl2, Wl2T, bl1p, bl2p);
  prep_graph<<<NGRAPH, 32, 0, stream>>>(x, ei, W1, b1, Gf16);
  gcn_main<<<NGRAPH / 8, 256, 0, stream>>>(Gf16, W2T3, Wl1T, Wl2T, b2p, bl1p, bl2p, out);
}
